// OVOLinearSemanticExtractor_75617194213983
// MI455X (gfx1250) — matrix operand order repaired and checked by reading, not compiled, not run on hardware
//
#include <hip/hip_runtime.h>

// ---------------------------------------------------------------------------
// OVO semantic extractor for MI455X (gfx1250, wave32).
//
// Roofline: ~8.3 GFLOP over 251 MB of f32 input -> stage3 is ~33 FLOP/byte,
// which needs matrix-core rates to ride the 23.3 TB/s HBM roof. So the 1x1
// convs run as v_wmma_f32_16x16x32_f16 GEMMs (f32 loads, f16 operands, f32
// accum). Weights live in LDS pre-packed in WMMA-A fragment order (f16), so
// the inner loop is: 2x ds_load_b128 (A) + 16 immediate-offset global b32 (B)
// + cvt + 10 WMMAs per K-step. All intermediate logits (~184 MB) fit the
// 192 MB L2, so the fused upsample+vote kernel gathers straight from them.
// ---------------------------------------------------------------------------

typedef __attribute__((ext_vector_type(16))) _Float16 v16h;
typedef __attribute__((ext_vector_type(8)))  float    v8f;

#define N_OVO   66
#define N_CLASS 12

// ---------------------------------------------------------------------------
// out[66, 8*HW] = w[66, C] * x[C, 8*HW] + b
// One wave: 32 pixels (2 N-subtiles) x all 5 M-tiles (80 rows, >=66 masked).
// LDS: (C/32)*5 A-fragments, each 32 lanes x 16 halfs (32 B), f16, pre-masked.
// ---------------------------------------------------------------------------
template<int C, int HW>
__global__ __launch_bounds__(128) void conv1x1_wmma_kernel(
    const float* __restrict__ x,    // [8, C, HW]
    const float* __restrict__ w,    // [66, C]
    const float* __restrict__ bias, // [66]
    float* __restrict__ out)        // [8, 66, HW]
{
    extern __shared__ __align__(32) _Float16 smem[];   // (C/32)*5*32*16 halfs

    const int lane = threadIdx.x & 31;
    const int m    = lane & 15;
    const int hi   = lane >> 4;

    // ---- Cooperative fill: pack weights as f16 WMMA-A fragments ----------
    // Fragment f = kbIdx*5 + mt; lane chunk = 16 halfs:
    //   K = kb + hi*16 + e (e=0..15)
    constexpr int NFRAG_LANES = (C / 32) * 5 * 32;
    for (int f = threadIdx.x; f < NFRAG_LANES; f += 128) {
        const int l2  = f & 31;
        const int fi  = f >> 5;
        const int mt  = fi % 5;
        const int kb  = (fi / 5) * 32;
        const int mm  = l2 & 15;
        const int h2  = l2 >> 4;
        const int o   = mt * 16 + mm;
        v16h af;
        if (o < N_OVO) {
            const float* wr = w + (size_t)o * C + kb + h2 * 16;
            #pragma unroll
            for (int e = 0; e < 8; ++e) af[e]     = (_Float16)wr[e];
            #pragma unroll
            for (int e = 0; e < 8; ++e) af[8 + e] = (_Float16)wr[8 + e];
        } else {
            #pragma unroll
            for (int e = 0; e < 16; ++e) af[e] = (_Float16)0.0f;
        }
        *(v16h*)(smem + (size_t)f * 16) = af;
    }
    __syncthreads();

    // ---- Tile coordinates -------------------------------------------------
    const int wv      = threadIdx.x >> 5;
    const int gw      = blockIdx.x * 4 + wv;
    const int pixBase = gw * 32;              // HW % 32 == 0 -> no image straddle
    const int img     = pixBase / HW;
    const int off     = pixBase - img * HW;

    // B: lane (n=m, half=hi) holds K = hi*16 + e (e=0..15) for its pixel col.
    const float* colBase = x + (size_t)img * C * HW + (size_t)(hi * 16) * HW + off + m;

    // ---- Accumulators, bias-initialized ----------------------------------
    v8f acc[5][2];
    #pragma unroll
    for (int mt = 0; mt < 5; ++mt) {
        #pragma unroll
        for (int r = 0; r < 8; ++r) {
            const int o = mt * 16 + hi * 8 + r;
            const float bv = (o < N_OVO) ? bias[o] : 0.0f;
            acc[mt][0][r] = bv;
            acc[mt][1][r] = bv;
        }
    }

    // ---- Main K loop ------------------------------------------------------
    for (int kbIdx = 0; kbIdx < C / 32; ++kbIdx) {
        const int kb = kbIdx * 32;

        v16h bf[2];
        #pragma unroll
        for (int st = 0; st < 2; ++st) {
            const float* p = colBase + (size_t)kb * HW + st * 16;
            #pragma unroll
            for (int e = 0; e < 16; ++e)          // immediate offsets: e*HW*4
                bf[st][e] = (_Float16)p[(size_t)e * HW];
        }

        #pragma unroll
        for (int mt = 0; mt < 5; ++mt) {
            const v16h af = *(const v16h*)(smem + ((size_t)(kbIdx * 5 + mt) * 32 + lane) * 16);
            acc[mt][0] = __builtin_amdgcn_wmma_f32_16x16x32_f16(
                false, af, false, bf[0], (short)0, acc[mt][0], false, false);
            acc[mt][1] = __builtin_amdgcn_wmma_f32_16x16x32_f16(
                false, af, false, bf[1], (short)0, acc[mt][1], false, false);
        }
    }

    // ---- Store logits [img, o, pix] (guard o < 66) ------------------------
    #pragma unroll
    for (int mt = 0; mt < 5; ++mt) {
        #pragma unroll
        for (int r = 0; r < 8; ++r) {
            const int o = mt * 16 + hi * 8 + r;
            if (o < N_OVO) {
                float* op = out + ((size_t)img * N_OVO + o) * HW + off + m;
                op[0]  = acc[mt][0][r];
                op[16] = acc[mt][1][r];
            }
        }
    }
}

// ---------------------------------------------------------------------------
// Fused bilinear-upsample + sum + one-vs-one vote.
// ---------------------------------------------------------------------------
__device__ __forceinline__ void hp_coord(int d, int H, float scale,
                                         int& i0, int& i1, float& f)
{
    const float src = ((float)d + 0.5f) * scale - 0.5f;
    const float fl  = floorf(src);
    f = src - fl;
    const int i = (int)fl;
    i0 = (i < 0) ? 0 : i;
    i1 = (i + 1 < H) ? (i + 1) : (H - 1);
}

__device__ __forceinline__ float bilin(const float* __restrict__ p, int W,
                                       int y0, int y1, int x0, int x1,
                                       float fy, float fx)
{
    const float a = p[y0 * W + x0];
    const float b = p[y0 * W + x1];
    const float c = p[y1 * W + x0];
    const float d = p[y1 * W + x1];
    const float top = a + fx * (b - a);
    const float bot = c + fx * (d - c);
    return top + fy * (bot - top);
}

__global__ __launch_bounds__(256) void vote_kernel(
    const float* __restrict__ l0,   // [8,66, 32, 32]
    const float* __restrict__ l1,   // [8,66, 64, 64]
    const float* __restrict__ l2,   // [8,66,128,128]
    const float* __restrict__ l3,   // [8,66,256,256]
    float* __restrict__ out)        // [8,12,256,256]
{
    const int idx = blockIdx.x * 256 + threadIdx.x;   // 8*256*256 threads
    const int xo = idx & 255;
    const int yo = (idx >> 8) & 255;
    const int n  = idx >> 16;

    int y0a, y1a, x0a, x1a; float fya, fxa;   // stage0, H=32
    int y0b, y1b, x0b, x1b; float fyb, fxb;   // stage1, H=64
    int y0c, y1c, x0c, x1c; float fyc, fxc;   // stage2, H=128
    hp_coord(yo,  32, 0.125f, y0a, y1a, fya); hp_coord(xo,  32, 0.125f, x0a, x1a, fxa);
    hp_coord(yo,  64, 0.25f,  y0b, y1b, fyb); hp_coord(xo,  64, 0.25f,  x0b, x1b, fxb);
    hp_coord(yo, 128, 0.5f,   y0c, y1c, fyc); hp_coord(xo, 128, 0.5f,   x0c, x1c, fxc);

    const float* b0 = l0 + (size_t)n * N_OVO * 1024;
    const float* b1 = l1 + (size_t)n * N_OVO * 4096;
    const float* b2 = l2 + (size_t)n * N_OVO * 16384;
    const float* b3 = l3 + (size_t)n * N_OVO * 65536 + yo * 256 + xo;

    float votes[N_CLASS];
    #pragma unroll
    for (int c = 0; c < N_CLASS; ++c) votes[c] = 0.0f;

    int p = 0;
    #pragma unroll
    for (int i = 0; i < N_CLASS; ++i) {
        #pragma unroll
        for (int j = i + 1; j < N_CLASS; ++j) {
            float v = b3[(size_t)p * 65536];
            v += bilin(b0 + (size_t)p * 1024,   32, y0a, y1a, x0a, x1a, fya, fxa);
            v += bilin(b1 + (size_t)p * 4096,   64, y0b, y1b, x0b, x1b, fyb, fxb);
            v += bilin(b2 + (size_t)p * 16384, 128, y0c, y1c, x0c, x1c, fyc, fxc);
            if (v > 0.0f) votes[i] += 1.0f; else votes[j] += 1.0f;
            ++p;
        }
    }

    const size_t obase = (size_t)n * N_CLASS * 65536 + yo * 256 + xo;
    #pragma unroll
    for (int c = 0; c < N_CLASS; ++c)
        out[obase + (size_t)c * 65536] = votes[c];
}

// ---------------------------------------------------------------------------
// setup_inputs() dict order (interleaved per stage):
//   0:stage0 1:w0 2:b0 3:stage1 4:w1 5:b1 6:stage2 7:w2 8:b2
//   9:stage3 10:w3 11:b3 12:last_only (==1)
// ---------------------------------------------------------------------------
extern "C" void kernel_launch(void* const* d_in, const int* in_sizes, int n_in,
                              void* d_out, int out_size, void* d_ws, size_t ws_size,
                              hipStream_t stream)
{
    (void)in_sizes; (void)n_in; (void)out_size; (void)ws_size;

    const float* s0 = (const float*)d_in[0];
    const float* w0 = (const float*)d_in[1];
    const float* b0 = (const float*)d_in[2];
    const float* s1 = (const float*)d_in[3];
    const float* w1 = (const float*)d_in[4];
    const float* b1 = (const float*)d_in[5];
    const float* s2 = (const float*)d_in[6];
    const float* w2 = (const float*)d_in[7];
    const float* b2 = (const float*)d_in[8];
    const float* s3 = (const float*)d_in[9];
    const float* w3 = (const float*)d_in[10];
    const float* b3 = (const float*)d_in[11];

    float* l0 = (float*)d_ws;                         // 8*66*1024
    float* l1 = l0 + (size_t)8 * N_OVO * 1024;        // 8*66*4096
    float* l2 = l1 + (size_t)8 * N_OVO * 4096;        // 8*66*16384
    float* l3 = l2 + (size_t)8 * N_OVO * 16384;       // 8*66*65536

    // LDS bytes per block = (C/32)*5 fragments * 32 lanes * 32 B = C*160
    conv1x1_wmma_kernel<512,  1024><<<  64, 128, 512 * 160, stream>>>(s0, w0, b0, l0);
    conv1x1_wmma_kernel<256,  4096><<< 256, 128, 256 * 160, stream>>>(s1, w1, b1, l1);
    conv1x1_wmma_kernel<128, 16384><<<1024, 128, 128 * 160, stream>>>(s2, w2, b2, l2);
    conv1x1_wmma_kernel< 64, 65536><<<4096, 128,  64 * 160, stream>>>(s3, w3, b3, l3);

    vote_kernel<<<(8 * 256 * 256) / 256, 256, 0, stream>>>(l0, l1, l2, l3, (float*)d_out);
}
